// Block2x2DenseL2SSM_88845693485295
// MI455X (gfx1250) — matrix operand order repaired and checked by reading, not compiled, not run on hardware
//
#include <hip/hip_runtime.h>
#include <hip/hip_bf16.h>

// MI455X / gfx1250, wave32. bf16 WMMA (f32 accum) for the three 64-wide GEMMs,
// TDM for weight staging, async global->LDS for activation staging, f32 VALU
// for the diagonal complex recurrence.

typedef __bf16 bf16;
typedef __attribute__((ext_vector_type(16))) __bf16 v16bf;
typedef __attribute__((ext_vector_type(8)))  float  v8f;
typedef __attribute__((ext_vector_type(4)))  unsigned int u32x4;
typedef __attribute__((ext_vector_type(8)))  int i32x8;
typedef __attribute__((ext_vector_type(4)))  int i32x4;

#define BATCH   32
#define TLEN    8192
#define DIM     64
#define NPAIR   32
#define CHUNK   128
#define NCHUNK  (TLEN / CHUNK)     // 64
#define UF_STRIDE 68               // padded f32 LDS row stride (floats)
#define Z_STRIDE  68               // padded bf16 LDS row stride (bf16)

// ---- workspace layout (bytes) ----
#define WS_W1      0                                   // bf16 [64][64] = scale*K12_raw  (contiguous with W2,W3)
#define WS_W2      8192                                // bf16 [64][64] = scale*K21_raw
#define WS_W3      16384                               // bf16 [64][64] = scale*K22_raw
#define WS_LAMBDA  24576                               // float2 [32]
#define WS_LAML    25088                               // float2 [32]
#define WS_H       32768                               // float2 [B][C][32] chunk suffix states
#define WS_ZSTART  (WS_H + BATCH*NCHUNK*NPAIR*8)       // float2 [B][C][32] chunk start states

// ---------------- CDNA5 data movers ----------------

// TDM: 1-D bf16 copy global->LDS via tensor descriptor (D#, ISA 8.3/8.4).
// Issue from one wave; caller waits s_wait_tensorcnt.
__device__ __forceinline__ void tdm_load_bf16_1d(unsigned lds_addr, const void* gsrc, int nelem) {
    unsigned long long ga = (unsigned long long)(uintptr_t)gsrc;
    u32x4 g0;
    g0[0] = 1u;                                        // count=1, user descriptor
    g0[1] = (unsigned)__builtin_amdgcn_readfirstlane((int)lds_addr);
    g0[2] = (unsigned)(ga & 0xFFFFFFFFu);              // global_addr[31:0]
    g0[3] = (unsigned)((ga >> 32) & 0x01FFFFFFu) | (2u << 30);  // global_addr[56:32] | type=2
    i32x8 g1;
    g1[0] = (1 << 16);                                 // wg_mask=0 | data_size=1 (2 bytes)
    g1[1] = (int)(((unsigned)nelem & 0xFFFFu) << 16);  // tensor_dim0[15:0] -> bits 63:48
    g1[2] = (int)((((unsigned)nelem >> 16) & 0xFFFFu)  // tensor_dim0[31:16] -> bits 79:64
                  | (1u << 16));                       // tensor_dim1 = 1
    g1[3] = (int)(((unsigned)nelem & 0xFFFFu) << 16);  // tile_dim0 -> bits 127:112
    g1[4] = 0;                                         // tile_dim1=0, tile_dim2=0 (1-D tile)
    g1[5] = nelem;                                     // tensor_dim0_stride[31:0]
    g1[6] = 0;
    g1[7] = 0;
    i32x4 z4 = {0, 0, 0, 0};
#if __clang_major__ >= 23
    i32x8 z8 = {0, 0, 0, 0, 0, 0, 0, 0};
    __builtin_amdgcn_tensor_load_to_lds(g0, g1, z4, z4, z8, 0);
#else
    __builtin_amdgcn_tensor_load_to_lds(g0, g1, z4, z4, 0);
#endif
}

// Async global->LDS: stage one u chunk (CHUNK x 64 f32, padded rows) without a
// VGPR round-trip. Tracked by ASYNCcnt; each lane copies 16 b128 quads.
__device__ __forceinline__ void stage_u_async(const float* __restrict__ u, size_t ubase, float* uS) {
    int tid = threadIdx.x;
    unsigned lds_base = (unsigned)(uintptr_t)(void*)uS;
#pragma unroll
    for (int i = 0; i < 16; ++i) {
        int idx4 = i * 128 + tid;                        // 2048 float4 chunks
        int row = idx4 >> 4, q = idx4 & 15;              // 16 quads per 64-float row
        unsigned ldsa = lds_base + (unsigned)(row * (UF_STRIDE * 4) + q * 16);
        const float* g = u + ubase + (size_t)idx4 * 4;
        asm volatile("global_load_async_to_lds_b128 %0, %1, off"
                     :: "v"(ldsa), "v"(g) : "memory");
    }
    asm volatile("s_wait_asynccnt 0" ::: "memory");      // own wave's copies done
}

// ---------------- WMMA fragment helpers (ISA 7.12.2 layouts) ----------------

// 16-bit A 16x32 from f32 LDS (convert at load): lanes 0-15 -> K {0..7,16..23},
// lanes 16-31 -> K {8..15,24..31}; M = lane&15.
__device__ __forceinline__ v16bf load_frag_a_f32(const float* lds, int m0, int kbase, int stride) {
    int lane = threadIdx.x & 31;
    const float* p = lds + (m0 + (lane & 15)) * stride;
    int kb = kbase + ((lane >> 4) << 3);
    v16bf a;
#pragma unroll
    for (int j = 0; j < 8; ++j) a[j] = (bf16)p[kb + j];
#pragma unroll
    for (int j = 0; j < 8; ++j) a[8 + j] = (bf16)p[kb + 16 + j];
    return a;
}

// 16-bit A 16x32 from bf16 LDS (z sequence)
__device__ __forceinline__ v16bf load_frag_a_bf16(const bf16* lds, int m0, int kbase, int stride) {
    int lane = threadIdx.x & 31;
    const bf16* p = lds + (m0 + (lane & 15)) * stride;
    int kb = kbase + ((lane >> 4) << 3);
    v16bf a;
#pragma unroll
    for (int j = 0; j < 8; ++j) a[j] = p[kb + j];
#pragma unroll
    for (int j = 0; j < 8; ++j) a[8 + j] = p[kb + 16 + j];
    return a;
}

// 16-bit B 32x16: lane = N (mod 16); lanes 0-15 hold K {0..7,16..23}, 16-31 hold K {8..15,24..31}.
// Weights stored [n][64] so the 16 K's are contiguous bf16.
__device__ __forceinline__ v16bf load_frag_b(const bf16* w, int n0, int kbase) {
    int lane = threadIdx.x & 31;
    const bf16* p = w + (n0 + (lane & 15)) * DIM + kbase + ((lane >> 4) << 3);
    v16bf b;
#pragma unroll
    for (int e = 0; e < 16; ++e) b[e] = p[e < 8 ? e : e + 8];
    return b;
}

// f32 C/D 16x16: element r -> M = r + 8*(lane>=16), N = lane&15
__device__ __forceinline__ void store_acc_lds(float* out, v8f acc, int m0, int n0, int stride) {
    int lane = threadIdx.x & 31;
    int n = n0 + (lane & 15);
    int mb = m0 + ((lane >> 4) << 3);
#pragma unroll
    for (int r = 0; r < 8; ++r) out[(mb + r) * stride + n] = acc[r];
}

__device__ __forceinline__ v8f wmma_bf16(v16bf a, v16bf b, v8f c) {
    return __builtin_amdgcn_wmma_f32_16x16x32_bf16(false, a, false, b, (short)0, c, false, false);
}

__device__ __forceinline__ float block_reduce_sum(float v, float* red) {
    int tid = threadIdx.x;
    red[tid] = v; __syncthreads();
    for (int s = 64; s > 0; s >>= 1) { if (tid < s) red[tid] += red[tid + s]; __syncthreads(); }
    float t = red[0]; __syncthreads();
    return t;
}

// ---------------- Kernel 0: sigma / lambda / weight prep (1 WG) ----------------
__global__ __launch_bounds__(128)
void setup_kernel(const float* __restrict__ rho_raw, const float* __restrict__ theta,
                  const float* __restrict__ K12, const float* __restrict__ K21,
                  const float* __restrict__ K22, const float* __restrict__ u0,
                  unsigned char* __restrict__ ws)
{
    __shared__ float K[128 * 128];
    __shared__ float red[128];
    __shared__ float uvec[128];
    __shared__ float vv[128];
    int tid = threadIdx.x;        // 0..127, one row each

    if (tid < 64) {
        int p = tid >> 1;
        float rho = (1.0f / (1.0f + expf(-rho_raw[p]))) * (1.0f - 0.001f);
        float ct = cosf(theta[p]), st = sinf(theta[p]);
        for (int c = 0; c < 64; ++c) {
            float v = 0.0f;
            if (c == 2 * p)     v = (tid & 1) ?  rho * st :  rho * ct;
            if (c == 2 * p + 1) v = (tid & 1) ?  rho * ct : -rho * st;
            K[tid * 128 + c] = v;
        }
        for (int c = 0; c < 64; ++c) K[tid * 128 + 64 + c] = K12[tid * 64 + c];
    } else {
        int rr = tid - 64;
        for (int c = 0; c < 64; ++c) K[tid * 128 + c]      = K21[rr * 64 + c];
        for (int c = 0; c < 64; ++c) K[tid * 128 + 64 + c] = K22[rr * 64 + c];
    }

    float x  = u0[tid];
    float n0 = sqrtf(block_reduce_sum(x * x, red));
    uvec[tid] = x / (n0 + 1e-12f);
    __syncthreads();

    float acc = 0.0f;
    for (int r = 0; r < 128; ++r) acc += K[r * 128 + tid] * uvec[r];
    float nv = sqrtf(block_reduce_sum(acc * acc, red));
    vv[tid] = acc / (nv + 1e-12f);
    __syncthreads();

    acc = 0.0f;
    for (int c = 0; c < 128; ++c) acc += K[tid * 128 + c] * vv[c];
    float ss    = block_reduce_sum(acc * acc, red);
    float sigma = ss / (sqrtf(ss) + 1e-12f);
    sigma = fmaxf(sigma, 1e-12f);
    float scale = 1.0f / fmaxf(sigma, 1.0f);

    if (tid < NPAIR) {
        float rho = (1.0f / (1.0f + expf(-rho_raw[tid]))) * (1.0f - 0.001f);
        float ct = cosf(theta[tid]), st = sinf(theta[tid]);
        float a = scale * rho * ct, b = scale * rho * st;   // lambda = a + i b
        ((float2*)(ws + WS_LAMBDA))[tid] = make_float2(a, b);
        float lr = 1.0f, li = 0.0f;
        for (int t = 0; t < CHUNK; ++t) {
            float nr = lr * a - li * b, ni = lr * b + li * a;
            lr = nr; li = ni;
        }
        ((float2*)(ws + WS_LAML))[tid] = make_float2(lr, li);
    }

    bf16* W1 = (bf16*)(ws + WS_W1);
    bf16* W2 = (bf16*)(ws + WS_W2);
    bf16* W3 = (bf16*)(ws + WS_W3);
    for (int i = tid; i < DIM * DIM; i += 128) {
        W1[i] = (bf16)(scale * K12[i]);
        W2[i] = (bf16)(scale * K21[i]);
        W3[i] = (bf16)(scale * K22[i]);
    }
}

// --------- shared GEMM: Bu[CHUNK][64] = u_chunk @ W1(B-operand), 4 waves ---------
__device__ __forceinline__ void bu_gemm(const float* uS, float* BuS, const bf16* w1S)
{
    int wave = threadIdx.x >> 5;
#pragma unroll
    for (int mi = 0; mi < 2; ++mi) {
        int mt = wave * 2 + mi;           // 8 M-tiles across 4 waves
#pragma unroll
        for (int nt = 0; nt < 4; ++nt) {
            v8f acc = {};
#pragma unroll
            for (int kt = 0; kt < 2; ++kt) {
                v16bf a = load_frag_a_f32(uS, mt * 16, kt * 32, UF_STRIDE);
                v16bf b = load_frag_b(w1S, nt * 16, kt * 32);
                acc = wmma_bf16(a, b, acc);
            }
            store_acc_lds(BuS, acc, mt * 16, nt * 16, DIM);
        }
    }
}

// ---------------- Kernel 1: per-chunk suffix state h ----------------
__global__ __launch_bounds__(128)
void chunk_state_kernel(const float* __restrict__ u, unsigned char* __restrict__ ws)
{
    __shared__ __attribute__((aligned(16))) float uS[CHUNK * UF_STRIDE];
    __shared__ float BuS[CHUNK * DIM];
    __shared__ __attribute__((aligned(16))) bf16 w1S[DIM * DIM];
    int tid = threadIdx.x;
    int c = blockIdx.x, b = blockIdx.y;

    if (tid < 32) {   // wave 0: TDM fetch of W1 (8KB), scalar-issued DMA
        tdm_load_bf16_1d((unsigned)(uintptr_t)(void*)w1S, ws + WS_W1, DIM * DIM);
        __builtin_amdgcn_s_wait_tensorcnt(0);
    }
    size_t ubase = ((size_t)b * TLEN + (size_t)c * CHUNK) * DIM;
    stage_u_async(u, ubase, uS);
    __syncthreads();

    bu_gemm(uS, BuS, w1S);
    __syncthreads();

    if (tid < NPAIR) {                          // wave 0: 32 complex scans, 128 steps
        float2 l = ((const float2*)(ws + WS_LAMBDA))[tid];
        float hr = 0.0f, hi = 0.0f;
        for (int t = 0; t < CHUNK; ++t) {
            float br = BuS[t * DIM + 2 * tid], bi = BuS[t * DIM + 2 * tid + 1];
            float nr = l.x * hr - l.y * hi + br;
            float ni = l.y * hr + l.x * hi + bi;
            hr = nr; hi = ni;
        }
        ((float2*)(ws + WS_H))[((size_t)b * NCHUNK + c) * NPAIR + tid] = make_float2(hr, hi);
    }
}

// ---------------- Kernel 2: sequential chunk combine + z_last ----------------
__global__ __launch_bounds__(1024)
void combine_kernel(unsigned char* __restrict__ ws, float* __restrict__ out)
{
    int tid = threadIdx.x;             // 1024 = B * NPAIR
    int b = tid >> 5, p = tid & 31;
    float2 lL = ((const float2*)(ws + WS_LAML))[p];
    const float2* h  = (const float2*)(ws + WS_H);
    float2*       zs = (float2*)(ws + WS_ZSTART);
    float sr = 0.0f, si = 0.0f;
    for (int c = 0; c < NCHUNK; ++c) {
        size_t idx = ((size_t)b * NCHUNK + c) * NPAIR + p;
        zs[idx] = make_float2(sr, si);
        float2 hc = h[idx];
        float nr = lL.x * sr - lL.y * si + hc.x;
        float ni = lL.y * sr + lL.x * si + hc.y;
        sr = nr; si = ni;
    }
    size_t zoff = (size_t)BATCH * TLEN * DIM + (size_t)b * DIM + 2 * p;
    out[zoff] = sr; out[zoff + 1] = si;        // z_last tail of d_out
}

// ---------------- Kernel 3: replay scan + fused y = z@W2 + u@W3 ----------------
__global__ __launch_bounds__(128)
void output_kernel(const float* __restrict__ u, unsigned char* __restrict__ ws,
                   float* __restrict__ out)
{
    __shared__ __attribute__((aligned(16))) float uS[CHUNK * UF_STRIDE];
    __shared__ float BuS[CHUNK * DIM];
    __shared__ __attribute__((aligned(16))) bf16 zS[CHUNK * Z_STRIDE];
    __shared__ __attribute__((aligned(16))) bf16 wS[3 * DIM * DIM];   // W1 | W2 | W3
    int tid = threadIdx.x;
    int c = blockIdx.x, b = blockIdx.y;

    if (tid < 32) {   // one TDM for all three weight matrices (24KB contiguous)
        tdm_load_bf16_1d((unsigned)(uintptr_t)(void*)wS, ws + WS_W1, 3 * DIM * DIM);
        __builtin_amdgcn_s_wait_tensorcnt(0);
    }
    size_t ubase = ((size_t)b * TLEN + (size_t)c * CHUNK) * DIM;
    stage_u_async(u, ubase, uS);
    __syncthreads();

    const bf16* w1S = wS;
    const bf16* w2S = wS + DIM * DIM;
    const bf16* w3S = wS + 2 * DIM * DIM;

    bu_gemm(uS, BuS, w1S);
    __syncthreads();

    if (tid < NPAIR) {                          // replay local scan, record z_t (pre-update)
        float2 l  = ((const float2*)(ws + WS_LAMBDA))[tid];
        float2 s0 = ((const float2*)(ws + WS_ZSTART))[((size_t)b * NCHUNK + c) * NPAIR + tid];
        float sr = s0.x, si = s0.y;
        for (int t = 0; t < CHUNK; ++t) {
            zS[t * Z_STRIDE + 2 * tid]     = (bf16)sr;
            zS[t * Z_STRIDE + 2 * tid + 1] = (bf16)si;
            float br = BuS[t * DIM + 2 * tid], bi = BuS[t * DIM + 2 * tid + 1];
            float nr = l.x * sr - l.y * si + br;
            float ni = l.y * sr + l.x * si + bi;
            sr = nr; si = ni;
        }
    }
    __syncthreads();

    int wave = tid >> 5, lane = tid & 31;
#pragma unroll
    for (int mi = 0; mi < 2; ++mi) {
        int mt = wave * 2 + mi;
#pragma unroll
        for (int nt = 0; nt < 4; ++nt) {
            v8f acc = {};
#pragma unroll
            for (int kt = 0; kt < 2; ++kt) {
                acc = wmma_bf16(load_frag_a_bf16(zS, mt * 16, kt * 32, Z_STRIDE),
                                load_frag_b(w2S, nt * 16, kt * 32), acc);
                acc = wmma_bf16(load_frag_a_f32(uS, mt * 16, kt * 32, UF_STRIDE),
                                load_frag_b(w3S, nt * 16, kt * 32), acc);
            }
            int n  = nt * 16 + (lane & 15);
            int mb = mt * 16 + ((lane >> 4) << 3);
#pragma unroll
            for (int r = 0; r < 8; ++r)
                out[((size_t)b * TLEN + (size_t)c * CHUNK + mb + r) * DIM + n] = acc[r];
        }
    }
}

// ---------------- host entry ----------------
extern "C" void kernel_launch(void* const* d_in, const int* in_sizes, int n_in,
                              void* d_out, int out_size, void* d_ws, size_t ws_size,
                              hipStream_t stream)
{
    const float* u       = (const float*)d_in[0];
    const float* rho_raw = (const float*)d_in[1];
    const float* theta   = (const float*)d_in[2];
    const float* K12     = (const float*)d_in[3];
    const float* K21     = (const float*)d_in[4];
    const float* K22     = (const float*)d_in[5];
    const float* u0      = (const float*)d_in[6];
    unsigned char* ws = (unsigned char*)d_ws;
    float* out = (float*)d_out;

    setup_kernel<<<1, 128, 0, stream>>>(rho_raw, theta, K12, K21, K22, u0, ws);
    dim3 grid(NCHUNK, BATCH);
    chunk_state_kernel<<<grid, 128, 0, stream>>>(u, ws);
    combine_kernel<<<1, 1024, 0, stream>>>(ws, out);
    output_kernel<<<grid, 128, 0, stream>>>(u, ws, out);
}
